// STMoE_DistilGPT2_19731079758574
// MI455X (gfx1250) — matrix operand order repaired and checked by reading, not compiled, not run on hardware
//
#include <hip/hip_runtime.h>
#include <hip/hip_bf16.h>
#include <math.h>

#define D_MODEL 768
#define N_HEAD  12
#define HEAD_D  64
#define N_LAYER 6
#define VOCAB   50258
#define SEQ     1024
#define BATCH   2
#define TOK     (BATCH*SEQ)
#define N_EXP   8
#define CAP     256
#define EXP_HID 2048

typedef __attribute__((ext_vector_type(16))) __bf16 v16bf;
typedef __attribute__((ext_vector_type(8)))  __bf16 v8bf;
typedef __attribute__((ext_vector_type(8)))  float  v8f;

__device__ __forceinline__ float gelu_tanh_f(float x){
    float x3 = x*x*x;
    return 0.5f*x*(1.0f + tanhf(0.7978845608028654f*(x + 0.044715f*x3)));
}
__device__ __forceinline__ float gelu_erf_f(float x){
    return 0.5f*x*(1.0f + erff(x*0.7071067811865476f));
}

// ---------------------------------------------------------------------------
// WMMA GEMM with LDS staging: C[M,N] = act(A[M,K] @ B + bias) + resid
//   bTrans==0: B is [K,N] row-major;  bTrans==1: B is [N,K] row-major (B^T)
//   act: 0=none, 1=tanh-gelu
//   blockDim = 256 (8 waves as 4x2), tile BM=128 x BN=128 x BK=32
//   each wave computes a 32x64 register tile = 2x4 WMMA fragments (8 wmma/step)
//   grid = (M/128, ceil(N/128), nbatch);  M % 128 == 0, K % 32 == 0 required
// ---------------------------------------------------------------------------
#define ASTRIDE 40   // bf16 elements per LDS row (32 + pad), 80B = 16B aligned
#define BSTRIDE 40

__global__ void __launch_bounds__(256)
gemm_wmma_kernel(const float* __restrict__ A,
                 const float* __restrict__ Bm,
                 float* __restrict__ C,
                 const float* __restrict__ bias,
                 const float* __restrict__ resid,
                 int M, int N, int K, int bTrans, int act,
                 long long sA, long long sB, long long sC,
                 long long sBias)
{
    __shared__ __align__(16) __bf16 Abuf[128 * ASTRIDE];
    __shared__ __align__(16) __bf16 Bbuf[128 * BSTRIDE];

    long long bz = blockIdx.z;
    A  += bz * sA;
    Bm += bz * sB;
    C  += bz * sC;
    if (bias)  bias  += bz * sBias;
    if (resid) resid += bz * sC;

    int tid   = threadIdx.x;
    int wave  = tid >> 5;
    int waveM = wave >> 1;      // 0..3  (M sub-tile of 32 rows)
    int waveN = wave & 1;       // 0..1  (N sub-tile of 64 cols)
    int lane  = tid & 31;
    int half  = lane >> 4;      // 0 or 1
    int l16   = lane & 15;
    int m0 = blockIdx.x * 128;
    int n0 = blockIdx.y * 128;
    bool nfull = (n0 + 128 <= N);

    v8f acc[2][4] = {};

    for (int k0 = 0; k0 < K; k0 += 32) {
        __syncthreads();
        // ---- stage A tile 128x32 (f32 -> bf16), coalesced b128 loads ----
        #pragma unroll
        for (int it = 0; it < 4; ++it) {
            int idx = tid + it * 256;           // 0..1023
            int row = idx >> 3;                 // 0..127
            int kg  = (idx & 7) * 4;            // 0..28
            const float4 v = *(const float4*)(A + (size_t)(m0 + row) * K + k0 + kg);
            __bf16* dst = Abuf + row * ASTRIDE + kg;
            dst[0] = (__bf16)v.x; dst[1] = (__bf16)v.y;
            dst[2] = (__bf16)v.z; dst[3] = (__bf16)v.w;
        }
        // ---- stage B tile 32x128 as Bbuf[n][k] (k-contiguous per row) ----
        if (!bTrans) {
            if (nfull) {
                #pragma unroll
                for (int it = 0; it < 4; ++it) {
                    int idx  = tid + it * 256;  // 0..1023
                    int krow = idx >> 5;        // 0..31
                    int ng   = (idx & 31) * 4;  // 0..124
                    const float4 v = *(const float4*)(Bm + (size_t)(k0 + krow) * N + n0 + ng);
                    Bbuf[(ng + 0) * BSTRIDE + krow] = (__bf16)v.x;
                    Bbuf[(ng + 1) * BSTRIDE + krow] = (__bf16)v.y;
                    Bbuf[(ng + 2) * BSTRIDE + krow] = (__bf16)v.z;
                    Bbuf[(ng + 3) * BSTRIDE + krow] = (__bf16)v.w;
                }
            } else {
                #pragma unroll
                for (int it = 0; it < 16; ++it) {
                    int idx  = tid + it * 256;  // 0..4095
                    int krow = idx >> 7;        // 0..31
                    int n    = idx & 127;
                    float v = (n0 + n < N) ? Bm[(size_t)(k0 + krow) * N + n0 + n] : 0.0f;
                    Bbuf[n * BSTRIDE + krow] = (__bf16)v;
                }
            }
        } else {
            if (nfull) {
                #pragma unroll
                for (int it = 0; it < 4; ++it) {
                    int idx  = tid + it * 256;  // 0..1023
                    int nrow = idx >> 3;        // 0..127
                    int kg   = (idx & 7) * 4;   // 0..28
                    const float4 v = *(const float4*)(Bm + (size_t)(n0 + nrow) * K + k0 + kg);
                    __bf16* dst = Bbuf + nrow * BSTRIDE + kg;
                    dst[0] = (__bf16)v.x; dst[1] = (__bf16)v.y;
                    dst[2] = (__bf16)v.z; dst[3] = (__bf16)v.w;
                }
            } else {
                #pragma unroll
                for (int it = 0; it < 16; ++it) {
                    int idx  = tid + it * 256;  // 0..4095
                    int nrow = idx >> 5;        // 0..127
                    int k    = idx & 31;
                    float v = (n0 + nrow < N) ? Bm[(size_t)(n0 + nrow) * K + k0 + k] : 0.0f;
                    Bbuf[nrow * BSTRIDE + k] = (__bf16)v;
                }
            }
        }
        // prefetch next K tile into cache while we compute
        if (k0 + 32 < K) {
            __builtin_prefetch(A + (size_t)(m0 + (tid >> 3)) * K + k0 + 32 + (tid & 7) * 4, 0, 1);
            if (!bTrans) __builtin_prefetch(Bm + (size_t)(k0 + 32 + (tid >> 5)) * N + n0 + (tid & 31) * 4, 0, 1);
            else         __builtin_prefetch(Bm + (size_t)(n0 + (tid >> 3)) * K + k0 + 32 + (tid & 7) * 4, 0, 1);
        }
        __syncthreads();

        // ---- fragments from LDS + 8 WMMA per wave (2 M-frags x 4 N-frags) ----
        v16bf af[2];
        #pragma unroll
        for (int f = 0; f < 2; ++f) {
            const __bf16* ap = Abuf + (waveM * 32 + f * 16 + l16) * ASTRIDE;
            v8bf a0 = *(const v8bf*)(ap + half * 8);
            v8bf a1 = *(const v8bf*)(ap + 16 + half * 8);
            af[f] = __builtin_shufflevector(a0, a1, 0,1,2,3,4,5,6,7,8,9,10,11,12,13,14,15);
        }
        #pragma unroll
        for (int t = 0; t < 4; ++t) {
            const __bf16* bp = Bbuf + (waveN * 64 + t * 16 + l16) * BSTRIDE;
            v8bf b0 = *(const v8bf*)(bp + half * 8);
            v8bf b1 = *(const v8bf*)(bp + 16 + half * 8);
            v16bf bfv = __builtin_shufflevector(b0, b1, 0,1,2,3,4,5,6,7,8,9,10,11,12,13,14,15);
            #pragma unroll
            for (int f = 0; f < 2; ++f) {
                acc[f][t] = __builtin_amdgcn_wmma_f32_16x16x32_bf16(
                    false, af[f], false, bfv, (short)0, acc[f][t], false, false);
            }
        }
    }

    // ---- epilogue: each wave writes its 32x64 strip ----
    #pragma unroll
    for (int f = 0; f < 2; ++f) {
        int mrow = m0 + waveM * 32 + f * 16;
        #pragma unroll
        for (int t = 0; t < 4; ++t) {
            int n = n0 + waveN * 64 + t * 16 + l16;
            if (n >= N) continue;
            float bv = bias ? bias[n] : 0.0f;
            #pragma unroll
            for (int r = 0; r < 8; ++r) {
                int m = mrow + r + half * 8;
                float v = acc[f][t][r] + bv;
                if (act == 1) v = gelu_tanh_f(v);
                if (resid) v += resid[(size_t)m * N + n];
                C[(size_t)m * N + n] = v;
            }
        }
    }
}

// ---------------------------------------------------------------------------
// Elementwise / norm / attention / MoE helper kernels
// ---------------------------------------------------------------------------
__global__ void embed_kernel(const int* __restrict__ ids,
                             const float* __restrict__ wte,
                             const float* __restrict__ wpe,
                             float* __restrict__ x)
{
    size_t idx = (size_t)blockIdx.x * 256 + threadIdx.x;
    if (idx >= (size_t)TOK * D_MODEL) return;
    int t = (int)(idx / D_MODEL), d = (int)(idx % D_MODEL);
    int n = t % SEQ;
    x[idx] = wte[(size_t)ids[t] * D_MODEL + d] + wpe[(size_t)n * D_MODEL + d];
}

__global__ void layernorm_kernel(const float* __restrict__ x,
                                 const float* __restrict__ g,
                                 const float* __restrict__ b,
                                 float* __restrict__ y)
{
    __shared__ float red[256];
    int t = blockIdx.x, tid = threadIdx.x;
    const float* xr = x + (size_t)t * D_MODEL;
    float s = 0.0f;
    for (int i = tid; i < D_MODEL; i += 256) s += xr[i];
    red[tid] = s; __syncthreads();
    for (int o = 128; o > 0; o >>= 1) { if (tid < o) red[tid] += red[tid + o]; __syncthreads(); }
    float mean = red[0] / (float)D_MODEL; __syncthreads();
    float v = 0.0f;
    for (int i = tid; i < D_MODEL; i += 256) { float d = xr[i] - mean; v += d * d; }
    red[tid] = v; __syncthreads();
    for (int o = 128; o > 0; o >>= 1) { if (tid < o) red[tid] += red[tid + o]; __syncthreads(); }
    float rstd = rsqrtf(red[0] / (float)D_MODEL + 1e-5f);
    for (int i = tid; i < D_MODEL; i += 256)
        y[(size_t)t * D_MODEL + i] = (xr[i] - mean) * rstd * g[i] + b[i];
}

__global__ void rmsnorm_kernel(const float* __restrict__ x,
                               const float* __restrict__ g,
                               float* __restrict__ y)
{
    __shared__ float red[256];
    int t = blockIdx.x, tid = threadIdx.x;
    const float* xr = x + (size_t)t * D_MODEL;
    float s = 0.0f;
    for (int i = tid; i < D_MODEL; i += 256) s += xr[i] * xr[i];
    red[tid] = s; __syncthreads();
    for (int o = 128; o > 0; o >>= 1) { if (tid < o) red[tid] += red[tid + o]; __syncthreads(); }
    float nrm = sqrtf(red[0]);
    float sc = 27.712812921102035f / fmaxf(nrm, 1e-12f);   // sqrt(768)/max(n,1e-12)
    for (int i = tid; i < D_MODEL; i += 256)
        y[(size_t)t * D_MODEL + i] = xr[i] * sc * g[i];
}

// one block per (query token, head); blockDim = 256
__global__ void attn_kernel(const float* __restrict__ qkv, float* __restrict__ out)
{
    __shared__ float sq[HEAD_D];
    __shared__ float s[SEQ];
    __shared__ float red[256];
    int t = blockIdx.x, h = blockIdx.y, tid = threadIdx.x;
    int b = t / SEQ, n = t % SEQ;
    const float* qrow = qkv + (size_t)t * 2304 + h * HEAD_D;
    if (tid < HEAD_D) sq[tid] = qrow[tid];
    __syncthreads();
    int nk = n + 1;
    for (int k = tid; k < nk; k += 256) {
        const float* krow = qkv + (size_t)(b * SEQ + k) * 2304 + D_MODEL + h * HEAD_D;
        float d = 0.0f;
        #pragma unroll 8
        for (int j = 0; j < HEAD_D; ++j) d += sq[j] * krow[j];
        s[k] = d * 0.125f;
    }
    __syncthreads();
    float mx = -3.4e38f;
    for (int k = tid; k < nk; k += 256) mx = fmaxf(mx, s[k]);
    red[tid] = mx; __syncthreads();
    for (int o = 128; o > 0; o >>= 1) { if (tid < o) red[tid] = fmaxf(red[tid], red[tid + o]); __syncthreads(); }
    mx = red[0]; __syncthreads();
    float sm = 0.0f;
    for (int k = tid; k < nk; k += 256) { float e = __expf(s[k] - mx); s[k] = e; sm += e; }
    red[tid] = sm; __syncthreads();
    for (int o = 128; o > 0; o >>= 1) { if (tid < o) red[tid] += red[tid + o]; __syncthreads(); }
    float inv = 1.0f / red[0]; __syncthreads();
    int d = tid & 63, part = tid >> 6;          // 4 partial sums per dim
    float accv = 0.0f;
    for (int k = part; k < nk; k += 4)
        accv += s[k] * qkv[(size_t)(b * SEQ + k) * 2304 + 2 * D_MODEL + h * HEAD_D + d];
    red[tid] = accv; __syncthreads();
    if (part == 0) {
        float v = red[d] + red[64 + d] + red[128 + d] + red[192 + d];
        out[(size_t)t * D_MODEL + h * HEAD_D + d] = v * inv;
    }
}

// gated-gelu: out[r,j] = gelu_exact(h[r,2048+j]) * h[r,j] * mb[j]
__global__ void geglu_kernel(const float* __restrict__ h,
                             const float* __restrict__ mb,
                             float* __restrict__ out,
                             long long rows, int expertMode)
{
    long long idx = (long long)blockIdx.x * 256 + threadIdx.x;
    long long total = rows * EXP_HID;
    if (idx >= total) return;
    long long r = idx / EXP_HID;
    int j = (int)(idx % EXP_HID);
    const float* mbp = mb + (expertMode ? (size_t)(r / 512) * EXP_HID : 0);
    float a  = h[r * (2 * EXP_HID) + j];
    float gt = h[r * (2 * EXP_HID) + EXP_HID + j];
    out[idx] = gelu_erf_f(gt) * a * mbp[j];
}

__global__ void add_kernel(float* __restrict__ y, const float* __restrict__ x, long long n)
{
    long long idx = (long long)blockIdx.x * 256 + threadIdx.x;
    if (idx < n) y[idx] += x[idx];
}

__global__ void fill_kernel(float* __restrict__ p, long long n)
{
    long long idx = (long long)blockIdx.x * 256 + threadIdx.x;
    if (idx < n) p[idx] = 0.0f;
}

// routing: one block per batch; sequential capacity scan (matches jnp.cumsum)
__global__ void route_kernel(const float* __restrict__ gl,
                             int* __restrict__ se, int* __restrict__ sc,
                             float* __restrict__ sw, float* __restrict__ aux)
{
    if (threadIdx.x != 0) return;
    int b = blockIdx.x;
    int a0[N_EXP]; float rawsum[N_EXP], cnt1[N_EXP];
    for (int e = 0; e < N_EXP; ++e) { a0[e] = 0; rawsum[e] = 0.f; cnt1[e] = 0.f; }
    float zsum = 0.0f;
    for (int n = 0; n < SEQ; ++n) {
        int t = b * SEQ + n;
        const float* lg = gl + (size_t)t * N_EXP;
        float mx = lg[0];
        for (int e = 1; e < N_EXP; ++e) mx = fmaxf(mx, lg[e]);
        float ex[N_EXP], sm = 0.0f;
        for (int e = 0; e < N_EXP; ++e) { ex[e] = __expf(lg[e] - mx); sm += ex[e]; }
        float lse = mx + __logf(sm); zsum += lse * lse;
        float inv = 1.0f / sm, raw[N_EXP];
        for (int e = 0; e < N_EXP; ++e) { raw[e] = ex[e] * inv; rawsum[e] += raw[e]; }
        int i0 = 0;
        for (int e = 1; e < N_EXP; ++e) if (raw[e] > raw[i0]) i0 = e;
        int i1 = (i0 == 0) ? 1 : 0;
        for (int e = 0; e < N_EXP; ++e) if (e != i0 && raw[e] > raw[i1]) i1 = e;
        cnt1[i0] += 1.0f;
        float g0 = raw[i0], g1 = raw[i1];
        float gs = fmaxf(g0 + g1, 1e-9f);
        float gn0 = g0 / gs, gn1 = g1 / gs;
        se[t * 2 + 0] = i0; se[t * 2 + 1] = i1;
        a0[i0] += 1;
        if (a0[i0] <= CAP) { sc[t * 2 + 0] = a0[i0] - 1; sw[t * 2 + 0] = gn0; }
        else               { sc[t * 2 + 0] = -1;          sw[t * 2 + 0] = 0.f; }
        sc[t * 2 + 1] = -1;
        sw[t * 2 + 1] = (gn1 > 0.2f) ? gn1 : -1.0f;       // threshold route
    }
    int prev[N_EXP], a1[N_EXP];
    for (int e = 0; e < N_EXP; ++e) { prev[e] = (a0[e] < CAP) ? a0[e] : CAP; a1[e] = 0; }
    for (int n = 0; n < SEQ; ++n) {
        int t = b * SEQ + n;
        float w1v = sw[t * 2 + 1];
        if (w1v >= 0.0f) {
            int e = se[t * 2 + 1];
            a1[e] += 1;
            int pos = a1[e] + prev[e];
            if (pos <= CAP) sc[t * 2 + 1] = pos - 1;
        }
    }
    float bal = 0.0f;
    for (int e = 0; e < N_EXP; ++e)
        bal += (rawsum[e] / (float)SEQ) * (cnt1[e] / (float)SEQ);
    atomicAdd(aux + 0, bal);
    atomicAdd(aux + 1, zsum);
}

// gather routed tokens: xd[e,b,c,:] = src[t,:]
__global__ void gather_kernel(const float* __restrict__ src,
                              const int* __restrict__ se, const int* __restrict__ sc,
                              float* __restrict__ xd)
{
    int s = blockIdx.x;
    int c = sc[s];
    if (c < 0) return;
    int t = s >> 1, e = se[s], b = t / SEQ;
    const float* sr = src + (size_t)t * D_MODEL;
    float* dst = xd + (((size_t)e * BATCH + b) * CAP + c) * D_MODEL;
    for (int d = threadIdx.x; d < D_MODEL; d += 256) dst[d] = sr[d];
}

// combine: mo[t,:] = sum_k w_k * eo[e_k,b,c_k,:]
__global__ void combine_kernel(const float* __restrict__ eo,
                               const int* __restrict__ se, const int* __restrict__ sc,
                               const float* __restrict__ sw, float* __restrict__ mo)
{
    int t = blockIdx.x, b = t / SEQ;
    for (int d = threadIdx.x; d < D_MODEL; d += 256) {
        float v = 0.0f;
        for (int k = 0; k < 2; ++k) {
            int c = sc[t * 2 + k];
            if (c >= 0) {
                int e = se[t * 2 + k];
                v += sw[t * 2 + k] * eo[(((size_t)e * BATCH + b) * CAP + c) * D_MODEL + d];
            }
        }
        mo[(size_t)t * D_MODEL + d] = v;
    }
}

__global__ void aux_kernel(const float* __restrict__ aux, float* __restrict__ out)
{
    // per layer: 0.01 * (E*E/(B*E)) * bal + 0.001 * z/(B*N); summed over 2 MoE layers
    out[0] = 0.01f * ((float)N_EXP / (float)BATCH) * (aux[0] + aux[2])
           + (0.001f / (float)TOK) * (aux[1] + aux[3]);
}

// ---------------------------------------------------------------------------
extern "C" void kernel_launch(void* const* d_in, const int* in_sizes, int n_in,
                              void* d_out, int out_size, void* d_ws, size_t ws_size,
                              hipStream_t stream)
{
    (void)in_sizes; (void)n_in; (void)out_size; (void)ws_size;
    const int*   ids      = (const int*)  d_in[0];
    const float* wte      = (const float*)d_in[2];
    const float* wpe      = (const float*)d_in[3];
    const float* ln1_g    = (const float*)d_in[4];
    const float* ln1_b    = (const float*)d_in[5];
    const float* wqkv     = (const float*)d_in[6];
    const float* bqkv     = (const float*)d_in[7];
    const float* wo       = (const float*)d_in[8];
    const float* bo       = (const float*)d_in[9];
    const float* ln2_g    = (const float*)d_in[10];
    const float* ln2_b    = (const float*)d_in[11];
    const float* mlp_wfc  = (const float*)d_in[12];
    const float* mlp_bfc  = (const float*)d_in[13];
    const float* mlp_wproj= (const float*)d_in[14];
    const float* mlp_bproj= (const float*)d_in[15];
    const float* lnf_g    = (const float*)d_in[16];
    const float* lnf_b    = (const float*)d_in[17];
    const float* gate_w   = (const float*)d_in[18];
    const float* exp_w1   = (const float*)d_in[19];
    const float* exp_b1   = (const float*)d_in[20];
    const float* exp_mb   = (const float*)d_in[21];
    const float* exp_w2   = (const float*)d_in[22];
    const float* exp_b2   = (const float*)d_in[23];
    const float* moe_rms  = (const float*)d_in[24];
    const float* ffb_rms  = (const float*)d_in[25];
    const float* ffb_w1   = (const float*)d_in[26];
    const float* ffb_b1   = (const float*)d_in[27];
    const float* ffb_mb   = (const float*)d_in[28];
    const float* ffb_w2   = (const float*)d_in[29];
    const float* ffb_b2   = (const float*)d_in[30];
    const float* ffa_rms  = (const float*)d_in[31];
    const float* ffa_w1   = (const float*)d_in[32];
    const float* ffa_b1   = (const float*)d_in[33];
    const float* ffa_mb   = (const float*)d_in[34];
    const float* ffa_w2   = (const float*)d_in[35];
    const float* ffa_b2   = (const float*)d_in[36];
    float* out = (float*)d_out;

    // workspace layout (f32, 256B aligned)
    char*  ws  = (char*)d_ws;
    size_t off = 0;
    auto alloc = [&](size_t elems) -> float* {
        float* p = (float*)(ws + off);
        off += ((elems * sizeof(float) + 255) / 256) * 256;
        return p;
    };
    float* X   = alloc((size_t)TOK * D_MODEL);
    float* Hb  = alloc((size_t)TOK * D_MODEL);
    float* RN  = alloc((size_t)TOK * D_MODEL);
    float* QKV = alloc((size_t)TOK * 3 * D_MODEL);
    float* ATT = alloc((size_t)TOK * D_MODEL);
    float* FFH = alloc((size_t)TOK * 2 * EXP_HID);     // also dense-MLP hidden (T x 3072)
    float* FFG = alloc((size_t)TOK * EXP_HID);
    float* SX  = alloc((size_t)TOK * D_MODEL);
    float* MO  = alloc((size_t)TOK * D_MODEL);
    float* GL  = alloc((size_t)TOK * N_EXP);
    float* XD  = alloc((size_t)N_EXP * BATCH * CAP * D_MODEL);
    float* EH  = alloc((size_t)N_EXP * BATCH * CAP * 2 * EXP_HID);
    float* EG  = alloc((size_t)N_EXP * BATCH * CAP * EXP_HID);
    float* EO  = alloc((size_t)N_EXP * BATCH * CAP * D_MODEL);
    float* SW  = alloc((size_t)TOK * 2);
    int*   SE  = (int*)alloc((size_t)TOK * 2);
    int*   SC  = (int*)alloc((size_t)TOK * 2);
    float* AUX = alloc(8);

    auto gemm = [&](const float* A, const float* Bm, float* C,
                    const float* bias, const float* resid,
                    int M, int N, int K, int bT, int act, int nb,
                    long long sA, long long sB, long long sC, long long sBias) {
        dim3 g(M / 128, (N + 127) / 128, nb);
        gemm_wmma_kernel<<<g, dim3(256), 0, stream>>>(A, Bm, C, bias, resid,
                                                      M, N, K, bT, act, sA, sB, sC, sBias);
    };

    fill_kernel<<<1, 256, 0, stream>>>(AUX, 8);
    embed_kernel<<<(TOK * D_MODEL + 255) / 256, 256, 0, stream>>>(ids, wte, wpe, X);

    for (int i = 0; i < N_LAYER; ++i) {
        // ---- attention ----
        layernorm_kernel<<<TOK, 256, 0, stream>>>(X, ln1_g + i * D_MODEL, ln1_b + i * D_MODEL, Hb);
        gemm(Hb, wqkv + (size_t)i * D_MODEL * 3 * D_MODEL, QKV,
             bqkv + (size_t)i * 3 * D_MODEL, nullptr,
             TOK, 3 * D_MODEL, D_MODEL, 0, 0, 1, 0, 0, 0, 0);
        attn_kernel<<<dim3(TOK, N_HEAD), 256, 0, stream>>>(QKV, ATT);
        gemm(ATT, wo + (size_t)i * D_MODEL * D_MODEL, X,
             bo + (size_t)i * D_MODEL, X,
             TOK, D_MODEL, D_MODEL, 0, 0, 1, 0, 0, 0, 0);

        layernorm_kernel<<<TOK, 256, 0, stream>>>(X, ln2_g + i * D_MODEL, ln2_b + i * D_MODEL, Hb);

        if (i == 0 || i == 3) {
            // ---- sparse MoE block on Hb ----
            int ls = (i == 0) ? 0 : 1;
            // expert_ff(rms(h, ffb)) + h
            rmsnorm_kernel<<<TOK, 256, 0, stream>>>(Hb, ffb_rms, RN);
            gemm(RN, ffb_w1, FFH, ffb_b1, nullptr, TOK, 2 * EXP_HID, D_MODEL, 0, 0, 1, 0, 0, 0, 0);
            geglu_kernel<<<((long long)TOK * EXP_HID + 255) / 256, 256, 0, stream>>>(FFH, ffb_mb, FFG, TOK, 0);
            gemm(FFG, ffb_w2, SX, ffb_b2, Hb, TOK, D_MODEL, EXP_HID, 0, 0, 1, 0, 0, 0, 0);
            // moe(rms(x, moe_rms))
            rmsnorm_kernel<<<TOK, 256, 0, stream>>>(SX, moe_rms, RN);
            gemm(RN, gate_w, GL, nullptr, nullptr, TOK, N_EXP, D_MODEL, 0, 0, 1, 0, 0, 0, 0);
            route_kernel<<<BATCH, 32, 0, stream>>>(GL, SE, SC, SW, AUX + ls * 2);
            fill_kernel<<<((long long)N_EXP * BATCH * CAP * D_MODEL + 255) / 256, 256, 0, stream>>>(
                XD, (long long)N_EXP * BATCH * CAP * D_MODEL);
            gather_kernel<<<TOK * 2, 256, 0, stream>>>(RN, SE, SC, XD);
            gemm(XD, exp_w1, EH, exp_b1, nullptr,
                 BATCH * CAP, 2 * EXP_HID, D_MODEL, 0, 0, N_EXP,
                 (long long)BATCH * CAP * D_MODEL, (long long)D_MODEL * 2 * EXP_HID,
                 (long long)BATCH * CAP * 2 * EXP_HID, 2 * EXP_HID);
            geglu_kernel<<<((long long)N_EXP * BATCH * CAP * EXP_HID + 255) / 256, 256, 0, stream>>>(
                EH, exp_mb, EG, (long long)N_EXP * BATCH * CAP, 1);
            gemm(EG, exp_w2, EO, exp_b2, nullptr,
                 BATCH * CAP, D_MODEL, EXP_HID, 0, 0, N_EXP,
                 (long long)BATCH * CAP * EXP_HID, (long long)EXP_HID * D_MODEL,
                 (long long)BATCH * CAP * D_MODEL, D_MODEL);
            combine_kernel<<<TOK, 256, 0, stream>>>(EO, SE, SC, SW, MO);
            add_kernel<<<((long long)TOK * D_MODEL + 255) / 256, 256, 0, stream>>>(SX, MO, (long long)TOK * D_MODEL);
            // expert_ff(rms(x, ffa)) + x
            rmsnorm_kernel<<<TOK, 256, 0, stream>>>(SX, ffa_rms, RN);
            gemm(RN, ffa_w1, FFH, ffa_b1, nullptr, TOK, 2 * EXP_HID, D_MODEL, 0, 0, 1, 0, 0, 0, 0);
            geglu_kernel<<<((long long)TOK * EXP_HID + 255) / 256, 256, 0, stream>>>(FFH, ffa_mb, FFG, TOK, 0);
            gemm(FFG, ffa_w2, SX, ffa_b2, SX, TOK, D_MODEL, EXP_HID, 0, 0, 1, 0, 0, 0, 0);
            // x = x + block_output
            add_kernel<<<((long long)TOK * D_MODEL + 255) / 256, 256, 0, stream>>>(X, SX, (long long)TOK * D_MODEL);
        } else {
            // ---- dense GPT-2 MLP ----
            int j = (i == 1) ? 0 : (i == 2) ? 1 : (i == 4) ? 2 : 3;
            gemm(Hb, mlp_wfc + (size_t)j * D_MODEL * 4 * D_MODEL, FFH,
                 mlp_bfc + (size_t)j * 4 * D_MODEL, nullptr,
                 TOK, 4 * D_MODEL, D_MODEL, 0, 1 /*tanh-gelu*/, 1, 0, 0, 0, 0);
            gemm(FFH, mlp_wproj + (size_t)j * 4 * D_MODEL * D_MODEL, X,
                 mlp_bproj + (size_t)j * D_MODEL, X,
                 TOK, D_MODEL, 4 * D_MODEL, 0, 0, 1, 0, 0, 0, 0);
        }
    }

    // final LN + logits (x @ wte^T) + aux scalar
    layernorm_kernel<<<TOK, 256, 0, stream>>>(X, lnf_g, lnf_b, Hb);
    gemm(Hb, wte, out, nullptr, nullptr, TOK, VOCAB, D_MODEL, 1 /*B transposed*/, 0, 1, 0, 0, 0, 0);
    aux_kernel<<<1, 1, 0, stream>>>(AUX, out + (size_t)TOK * VOCAB);
}
